// QuantMSA_8375186227287
// MI455X (gfx1250) — 8-bit matrix operand order repaired and checked by reading, compiled, not run on hardware
//
#include <hip/hip_runtime.h>
#include <hip/hip_bf16.h>
#include <stdint.h>

#define Bsz   16
#define Tlen  512
#define Emb   768
#define Hn    12
#define Dh    64
#define N3    2304
#define BT    8192
#define BH    192

#define LDSK  80
#define BUFB  (64 * LDSK)

typedef __attribute__((ext_vector_type(8))) int v8i;
typedef int v4i_vs __attribute__((__vector_size__(16)));
typedef __attribute__((address_space(3))) signed char lds_char;
typedef __attribute__((address_space(3))) v4i_vs lds_v4;
typedef __attribute__((address_space(1))) v4i_vs glb_v4;

__device__ __forceinline__ float wave_max_f(float v) {
#pragma unroll
  for (int m = 16; m >= 1; m >>= 1) v = fmaxf(v, __shfl_xor(v, m, 32));
  return v;
}
__device__ __forceinline__ float wave_sum_f(float v) {
#pragma unroll
  for (int m = 16; m >= 1; m >>= 1) v += __shfl_xor(v, m, 32);
  return v;
}
__device__ __forceinline__ float scale_from_bits(const unsigned* p) {
  return __uint_as_float(*p) * (1.0f / 127.0f) + 1e-12f;
}

#if __has_builtin(__builtin_amdgcn_global_load_async_to_lds_b128)
#define HAVE_ASYNC_LDS 1
#else
#define HAVE_ASYNC_LDS 0
#endif

__device__ __forceinline__ void copy16_g2l(const signed char* g, lds_char* l) {
#if HAVE_ASYNC_LDS
  __builtin_amdgcn_global_load_async_to_lds_b128((glb_v4*)g, (lds_v4*)l, 0, 0);
#else
  *(lds_v4*)l = *(const v4i_vs*)g;
#endif
}
template <int N>
__device__ __forceinline__ void async_wait_le() {
#if HAVE_ASYNC_LDS
#if __has_builtin(__builtin_amdgcn_s_wait_asynccnt)
  __builtin_amdgcn_s_wait_asynccnt(N);
#else
  asm volatile("s_wait_asynccnt %0" ::"n"(N) : "memory");
#endif
#endif
}

__device__ __forceinline__ v8i load_a8(const signed char* base, int ldk, int lane) {
  const signed char* p = base + (size_t)(lane & 15) * ldk + ((lane >> 4) << 4);
  int2 c0 = *(const int2*)(p);
  int2 c1 = *(const int2*)(p + 8);
  int2 c2 = *(const int2*)(p + 32);
  int2 c3 = *(const int2*)(p + 40);
  v8i a;
  a[0] = c0.x; a[1] = c0.y; a[2] = c1.x; a[3] = c1.y;
  a[4] = c2.x; a[5] = c2.y; a[6] = c3.x; a[7] = c3.y;
  return a;
}
__device__ __forceinline__ v8i load_b8_lds(const lds_char* base, int lane) {
  const lds_char* p = base + (lane & 15) * LDSK + ((lane >> 4) << 4);
  v4i_vs c0 = *(const lds_v4*)(p);
  v4i_vs c1 = *(const lds_v4*)(p + 32);
  v8i b;
  b[0] = c0[0]; b[1] = c0[1]; b[2] = c0[2]; b[3] = c0[3];
  b[4] = c1[0]; b[5] = c1[1]; b[6] = c1[2]; b[7] = c1[3];
  return b;
}

template <bool SGNA>
__device__ __forceinline__ void wmma4(const lds_char* bbuf, int lane, v8i a,
                                      v8i (&acc)[4]) {
  v8i b0 = load_b8_lds(bbuf + 0 * 16 * LDSK, lane);
  v8i b1 = load_b8_lds(bbuf + 1 * 16 * LDSK, lane);
  v8i b2 = load_b8_lds(bbuf + 2 * 16 * LDSK, lane);
  v8i b3 = load_b8_lds(bbuf + 3 * 16 * LDSK, lane);
  acc[0] = __builtin_amdgcn_wmma_i32_16x16x64_iu8(SGNA, a, true, b0, acc[0], false, false);
  acc[1] = __builtin_amdgcn_wmma_i32_16x16x64_iu8(SGNA, a, true, b1, acc[1], false, false);
  acc[2] = __builtin_amdgcn_wmma_i32_16x16x64_iu8(SGNA, a, true, b2, acc[2], false, false);
  acc[3] = __builtin_amdgcn_wmma_i32_16x16x64_iu8(SGNA, a, true, b3, acc[3], false, false);
}

__global__ void init_scal_kernel(unsigned* scal) {
  if (threadIdx.x < 64) scal[threadIdx.x] = 0u;
}

__global__ void quant_weights_kernel(const float* __restrict__ W,
                                     signed char* __restrict__ Wi,
                                     float* __restrict__ sw, int K) {
  int n = blockIdx.x, tid = threadIdx.x;
  const float* wr = W + (size_t)n * K;
  float m = 0.f;
  for (int k = tid; k < K; k += 256) m = fmaxf(m, fabsf(wr[k]));
  m = wave_max_f(m);
  __shared__ float red[9];
  int w = tid >> 5, lane = tid & 31;
  if (lane == 0) red[w] = m;
  __syncthreads();
  if (tid == 0) {
    float mm = red[0];
    for (int i = 1; i < 8; ++i) mm = fmaxf(mm, red[i]);
    red[8] = mm * (1.0f / 127.0f);
  }
  __syncthreads();
  float s = red[8];
  if (tid == 0) sw[n] = s;
  float inv = 1.0f / (s + 1e-30f);
  for (int k = tid; k < K; k += 256)
    Wi[(size_t)n * K + k] = (signed char)fminf(fmaxf(rintf(wr[k] * inv), -128.f), 127.f);
}

__global__ void quant_x_kernel(const float* __restrict__ x, const float* s_x,
                               signed char* __restrict__ x8) {
  float inv = 1.0f / (*s_x);
  size_t n = (size_t)BT * Emb;
  for (size_t i = (size_t)blockIdx.x * blockDim.x + threadIdx.x; i < n;
       i += (size_t)gridDim.x * blockDim.x)
    x8[i] = (signed char)fminf(fmaxf(rintf(x[i] * inv), -128.f), 127.f);
}

__global__ void __launch_bounds__(256)
gemm_lin_kernel(const signed char* __restrict__ A,
                const signed char* __restrict__ Wi,
                const float* __restrict__ sw,
                const float* __restrict__ bias,
                const float* s_direct, const unsigned* s_bits,
                float* __restrict__ Y, unsigned* maxbits, int N, int K) {
  __shared__ __align__(16) signed char bsm[2 * BUFB];
  const int tid = threadIdx.x;
  const int lane = tid & 31;
  const int w = tid >> 5;
  const int row0 = blockIdx.y * 128 + w * 16;
  const int col0 = blockIdx.x * 64;
  const float s_in = s_direct ? *s_direct : scale_from_bits(s_bits);

  lds_char* lds0 = (lds_char*)&bsm[0];
  const int scol = tid >> 2, sch = (tid & 3) << 4;
  const signed char* gB = Wi + (size_t)(col0 + scol) * K + sch;
  lds_char* stDst = lds0 + scol * LDSK + sch;

  v8i zero = {0, 0, 0, 0, 0, 0, 0, 0};
  v8i acc[4];
#pragma unroll
  for (int t = 0; t < 4; ++t) acc[t] = zero;

  const signed char* Abase = A + (size_t)row0 * K;
  copy16_g2l(gB, stDst);
  v8i a_cur = load_a8(Abase, K, lane);

  int kk = 0;
  for (; kk + 64 < K; kk += 64) {
    const int cur = (kk >> 6) & 1;
    copy16_g2l(gB + kk + 64, stDst + (1 - cur) * BUFB);
    async_wait_le<1>();
    __syncthreads();
    v8i a = a_cur;
    a_cur = load_a8(Abase + kk + 64, K, lane);
    wmma4<true>(lds0 + cur * BUFB, lane, a, acc);
    __syncthreads();
  }
  async_wait_le<0>();
  __syncthreads();
  wmma4<true>(lds0 + ((kk >> 6) & 1) * BUFB, lane, a_cur, acc);

  const int nl = lane & 15, mb = (lane >> 4) << 3;
  float lmax = 0.f;
#pragma unroll
  for (int t = 0; t < 4; ++t) {
    int col = col0 + t * 16 + nl;
    float sacc = s_in * sw[col] + 1e-30f;
    float bint = rintf(bias[col] / sacc);
#pragma unroll
    for (int r = 0; r < 8; ++r) {
      float y = ((float)acc[t][r] + bint) * sacc;
      Y[(size_t)(row0 + mb + r) * N + col] = y;
      lmax = fmaxf(lmax, fabsf(y));
    }
  }
  lmax = wave_max_f(lmax);
  if (lane == 0) atomicMax(maxbits, __float_as_uint(lmax));
}

__global__ void __launch_bounds__(256)
gemm_qk_kernel(const signed char* __restrict__ qkv8, float* __restrict__ qkf,
               const unsigned* sbits_qkv, unsigned* maxbits) {
  __shared__ __align__(16) signed char bsm[BUFB];
  const int tid = threadIdx.x;
  const int lane = tid & 31;
  const int w = tid >> 5;
  const int bh = blockIdx.z, b = bh / Hn, h = bh % Hn;
  const int row0 = blockIdx.y * 128 + w * 16;
  const int col0 = blockIdx.x * 64;
  float sq = scale_from_bits(sbits_qkv);
  float sc = sq * sq * 0.125f;

  lds_char* lds0 = (lds_char*)&bsm[0];
  const int scol = tid >> 2, sch = (tid & 3) << 4;
  copy16_g2l(qkv8 + (size_t)(b * Tlen + col0 + scol) * N3 + Emb + h * Dh + sch,
             lds0 + scol * LDSK + sch);

  const signed char* qbase = qkv8 + (size_t)(b * Tlen + row0) * N3 + h * Dh;
  v8i a = load_a8(qbase, N3, lane);
  async_wait_le<0>();
  __syncthreads();

  v8i zero = {0, 0, 0, 0, 0, 0, 0, 0};
  v8i acc[4];
#pragma unroll
  for (int t = 0; t < 4; ++t) acc[t] = zero;
  wmma4<true>(lds0, lane, a, acc);

  const int nl = lane & 15, mb = (lane >> 4) << 3;
  float lmax = 0.f;
#pragma unroll
  for (int t = 0; t < 4; ++t) {
    int col = col0 + t * 16 + nl;
#pragma unroll
    for (int r = 0; r < 8; ++r) {
      float y = (float)acc[t][r] * sc;
      qkf[((size_t)bh * Tlen + row0 + mb + r) * Tlen + col] = y;
      lmax = fmaxf(lmax, fabsf(y));
    }
  }
  lmax = wave_max_f(lmax);
  if (lane == 0) atomicMax(maxbits, __float_as_uint(lmax));
}

__global__ void requant_s8_kernel(const float* __restrict__ y,
                                  signed char* __restrict__ q, size_t n,
                                  const unsigned* maxbits) {
  float inv = 1.0f / scale_from_bits(maxbits);
  for (size_t i = (size_t)blockIdx.x * blockDim.x + threadIdx.x; i < n;
       i += (size_t)gridDim.x * blockDim.x)
    q[i] = (signed char)fminf(fmaxf(rintf(y[i] * inv), -128.f), 127.f);
}

__global__ void int_softmax_kernel(const signed char* __restrict__ qk8,
                                   unsigned short* __restrict__ amap16,
                                   const unsigned* qk_maxbits,
                                   unsigned* q16_max) {
  const int lane = threadIdx.x & 31;
  const int row = blockIdx.x * 8 + (threadIdx.x >> 5);
  const signed char* rp = qk8 + (size_t)row * Tlen;
  float s = scale_from_bits(qk_maxbits);

  int vi[16];
  int mx = -1000;
#pragma unroll
  for (int i = 0; i < 16; ++i) {
    vi[i] = rp[lane + (i << 5)];
    mx = vi[i] > mx ? vi[i] : mx;
  }
  float mf = wave_max_f((float)mx);
  float x0 = floorf(-0.6931f / s);
  float bI = floorf(2.70732486f / s);
  float cI = floorf(2.79213917f / (0.35815147f * s * s));
  float e[16], sum = 0.f;
#pragma unroll
  for (int i = 0; i < 16; ++i) {
    float xi = (float)vi[i] - mf;
    xi = fmaxf(xi, 30.f * x0);
    float q = floorf(xi / x0);
    float r = xi - x0 * q;
    float z = r * (r + bI) + cI;
    float ev = fmaxf(floorf(ldexpf(z, 30 - (int)q)), 0.f);
    e[i] = ev;
    sum += ev;
  }
  sum = wave_sum_f(sum);
  float inv = 32767.f / (sum + 1e-12f);
  float lmax = 0.f;
#pragma unroll
  for (int i = 0; i < 16; ++i) {
    float q16 = fminf(fmaxf(rintf(e[i] * inv), 0.f), 32767.f);
    amap16[(size_t)row * Tlen + lane + (i << 5)] = (unsigned short)q16;
    lmax = fmaxf(lmax, q16);
  }
  lmax = wave_max_f(lmax);
  if (lane == 0) atomicMax(q16_max, (unsigned)lmax);
}

__global__ void requant_amap_kernel(const unsigned short* __restrict__ a16,
                                    unsigned char* __restrict__ a8, size_t n,
                                    const unsigned* q16max) {
  float s_amap = (float)(*q16max) * (1.f / 32767.f) * (1.f / 127.f) + 1e-12f;
  float inv = (1.f / 32767.f) / s_amap;
  for (size_t i = (size_t)blockIdx.x * blockDim.x + threadIdx.x; i < n;
       i += (size_t)gridDim.x * blockDim.x)
    a8[i] = (unsigned char)fminf(fmaxf(rintf((float)a16[i] * inv), 0.f), 127.f);
}

__global__ void transpose_v_kernel(const signed char* __restrict__ qkv8,
                                   signed char* __restrict__ vT) {
  size_t total = (size_t)BH * Dh * Tlen;
  for (size_t o = (size_t)blockIdx.x * blockDim.x + threadIdx.x; o < total;
       o += (size_t)gridDim.x * blockDim.x) {
    int t = (int)(o & (Tlen - 1));
    int d = (int)((o >> 9) & (Dh - 1));
    int bh = (int)(o >> 15);
    int b = bh / Hn, h = bh % Hn;
    vT[o] = qkv8[(size_t)(b * Tlen + t) * N3 + 2 * Emb + h * Dh + d];
  }
}

__global__ void __launch_bounds__(256)
gemm_av_kernel(const unsigned char* __restrict__ amap8,
               const signed char* __restrict__ vT,
               const unsigned* q16max, const unsigned* sbits_qkv,
               float* __restrict__ avf, unsigned* maxbits) {
  __shared__ __align__(16) signed char bsm[2 * BUFB];
  const int tid = threadIdx.x;
  const int lane = tid & 31;
  const int w = tid >> 5;
  const int bh = blockIdx.z;
  const int row0 = blockIdx.y * 128 + w * 16;

  lds_char* lds0 = (lds_char*)&bsm[0];
  const int scol = tid >> 2, sch = (tid & 3) << 4;
  const signed char* gB = vT + ((size_t)bh * Dh + scol) * Tlen + sch;
  lds_char* stDst = lds0 + scol * LDSK + sch;

  v8i zero = {0, 0, 0, 0, 0, 0, 0, 0};
  v8i acc[4];
#pragma unroll
  for (int t = 0; t < 4; ++t) acc[t] = zero;

  const signed char* abase =
      (const signed char*)amap8 + (size_t)(bh * Tlen + row0) * Tlen;
  copy16_g2l(gB, stDst);
  v8i a_cur = load_a8(abase, Tlen, lane);

  int kk = 0;
  for (; kk + 64 < Tlen; kk += 64) {
    const int cur = (kk >> 6) & 1;
    copy16_g2l(gB + kk + 64, stDst + (1 - cur) * BUFB);
    async_wait_le<1>();
    __syncthreads();
    v8i a = a_cur;
    a_cur = load_a8(abase + kk + 64, Tlen, lane);
    wmma4<false>(lds0 + cur * BUFB, lane, a, acc);
    __syncthreads();
  }
  async_wait_le<0>();
  __syncthreads();
  wmma4<false>(lds0 + ((kk >> 6) & 1) * BUFB, lane, a_cur, acc);

  float s_amap = (float)(*q16max) * (1.f / 32767.f) * (1.f / 127.f) + 1e-12f;
  float sc = s_amap * scale_from_bits(sbits_qkv);
  const int nl = lane & 15, mb = (lane >> 4) << 3;
  float lmax = 0.f;
#pragma unroll
  for (int t = 0; t < 4; ++t) {
    int col = t * 16 + nl;
#pragma unroll
    for (int r = 0; r < 8; ++r) {
      float y = (float)acc[t][r] * sc;
      avf[((size_t)bh * Tlen + row0 + mb + r) * Dh + col] = y;
      lmax = fmaxf(lmax, fabsf(y));
    }
  }
  lmax = wave_max_f(lmax);
  if (lane == 0) atomicMax(maxbits, __float_as_uint(lmax));
}

__global__ void requant_av_kernel(const float* __restrict__ avf,
                                  signed char* __restrict__ aout8,
                                  const unsigned* maxbits) {
  float inv = 1.0f / scale_from_bits(maxbits);
  size_t total = (size_t)BH * Tlen * Dh;
  for (size_t i = (size_t)blockIdx.x * blockDim.x + threadIdx.x; i < total;
       i += (size_t)gridDim.x * blockDim.x) {
    int d = (int)(i & (Dh - 1));
    int t = (int)((i >> 6) & (Tlen - 1));
    int bh = (int)(i >> 15);
    int b = bh / Hn, h = bh % Hn;
    float v = fminf(fmaxf(rintf(avf[i] * inv), -128.f), 127.f);
    aout8[(size_t)(b * Tlen + t) * Emb + h * Dh + d] = (signed char)v;
  }
}

__global__ void finalize_kernel(const float* __restrict__ y,
                                float* __restrict__ out,
                                const unsigned* maxbits) {
  float s = scale_from_bits(maxbits);
  float inv = 1.0f / s;
  size_t n = (size_t)BT * Emb;
  for (size_t i = (size_t)blockIdx.x * blockDim.x + threadIdx.x; i < n;
       i += (size_t)gridDim.x * blockDim.x)
    out[i] = fminf(fmaxf(rintf(y[i] * inv), -128.f), 127.f) * s;
  if (blockIdx.x == 0 && threadIdx.x == 0) out[n] = s;
}

extern "C" void kernel_launch(void* const* d_in, const int* in_sizes, int n_in,
                              void* d_out, int out_size, void* d_ws, size_t ws_size,
                              hipStream_t stream) {
  (void)in_sizes; (void)n_in; (void)out_size; (void)ws_size;
  const float* x_hat = (const float*)d_in[0];
  const float* s_x   = (const float*)d_in[1];
  const float* w_in  = (const float*)d_in[2];
  const float* b_in  = (const float*)d_in[3];
  const float* w_out = (const float*)d_in[4];
  const float* b_out = (const float*)d_in[5];
  float* out = (float*)d_out;

  char* ws = (char*)d_ws;
  auto al = [](size_t x) { return (x + 255) & ~(size_t)255; };
  size_t off = 0;
  unsigned* scal = (unsigned*)(ws + off);        off += al(1024);
  signed char* Wi_in  = (signed char*)(ws + off); off += al((size_t)N3 * Emb);
  float*       sw_in  = (float*)(ws + off);       off += al((size_t)N3 * 4);
  signed char* Wi_out = (signed char*)(ws + off); off += al((size_t)Emb * Emb);
  float*       sw_out = (float*)(ws + off);       off += al((size_t)Emb * 4);
  signed char* x8     = (signed char*)(ws + off); off += al((size_t)BT * Emb);
  signed char* qkv8   = (signed char*)(ws + off); off += al((size_t)BT * N3);
  signed char* vT     = (signed char*)(ws + off); off += al((size_t)BH * Dh * Tlen);
  signed char* aout8  = (signed char*)(ws + off); off += al((size_t)BT * Emb);
  char* arena1 = ws + off; off += al((size_t)BT * N3 * 4);
  char* arena2 = ws + off; off += al((size_t)BH * Tlen * Tlen * 4);

  float*          y_qkv  = (float*)arena1;
  signed char*    qk8    = (signed char*)arena1;
  unsigned char*  amap8  = (unsigned char*)arena1;
  float*          qkf    = (float*)arena2;
  unsigned short* amap16 = (unsigned short*)arena2;
  float*          avf    = (float*)arena2;
  float*          y_out  = (float*)arena2;

  unsigned* mx_qkv = scal + 0;
  unsigned* mx_qk  = scal + 1;
  unsigned* mx_p16 = scal + 2;
  unsigned* mx_av  = scal + 3;
  unsigned* mx_out = scal + 4;

  dim3 blk(256);
  init_scal_kernel<<<1, 64, 0, stream>>>(scal);
  quant_weights_kernel<<<N3, blk, 0, stream>>>(w_in, Wi_in, sw_in, Emb);
  quant_weights_kernel<<<Emb, blk, 0, stream>>>(w_out, Wi_out, sw_out, Emb);
  quant_x_kernel<<<2048, blk, 0, stream>>>(x_hat, s_x, x8);

  gemm_lin_kernel<<<dim3(N3 / 64, BT / 128), blk, 0, stream>>>(
      x8, Wi_in, sw_in, b_in, s_x, nullptr, y_qkv, mx_qkv, N3, Emb);
  requant_s8_kernel<<<4096, blk, 0, stream>>>(y_qkv, qkv8, (size_t)BT * N3, mx_qkv);
  transpose_v_kernel<<<2048, blk, 0, stream>>>(qkv8, vT);

  gemm_qk_kernel<<<dim3(Tlen / 64, Tlen / 128, BH), blk, 0, stream>>>(
      qkv8, qkf, mx_qkv, mx_qk);
  requant_s8_kernel<<<4096, blk, 0, stream>>>(qkf, qk8,
                                              (size_t)BH * Tlen * Tlen, mx_qk);

  int_softmax_kernel<<<(BH * Tlen) / 8, blk, 0, stream>>>(qk8, amap16, mx_qk, mx_p16);
  requant_amap_kernel<<<4096, blk, 0, stream>>>(amap16, amap8,
                                                (size_t)BH * Tlen * Tlen, mx_p16);

  gemm_av_kernel<<<dim3(1, Tlen / 128, BH), blk, 0, stream>>>(
      amap8, vT, mx_p16, mx_qkv, avf, mx_av);
  requant_av_kernel<<<2048, blk, 0, stream>>>(avf, aout8, mx_av);

  gemm_lin_kernel<<<dim3(Emb / 64, BT / 128), blk, 0, stream>>>(
      aout8, Wi_out, sw_out, b_out, nullptr, mx_av, y_out, mx_out, Emb, Emb);
  finalize_kernel<<<2048, blk, 0, stream>>>(y_out, out, mx_out);
}
